// CPIModel_with_FP_43765716746553
// MI455X (gfx1250) — matrix operand order repaired and checked by reading, not compiled, not run on hardware
//
#include <hip/hip_runtime.h>
#include <hip/hip_bf16.h>

// ---------------------------------------------------------------------------
// CPI MPNN + attention model for MI455X (gfx1250, wave32).
// All dense GEMMs run through v_wmma_f32_16x16x32_f16 (f16 inputs, f32 accum).
// Fast vectorized (b128) global->LDS staging for interior tiles; guarded
// scalar path only for edge tiles. BN / scatter / pool / attention-core /
// loss are scalar helper kernels. Workspace budget: ~440 MB of d_ws.
// ---------------------------------------------------------------------------

#define N_NODES 60000
#define N_EDGES 120000
#define NB      1024
#define DIN     88
#define DE      32
#define EMB     256
#define NL      4
#define HIDD    512
#define NH      8
#define HD      64

typedef __attribute__((ext_vector_type(16))) _Float16 v16h;
typedef __attribute__((ext_vector_type(8)))  float    v8f;

union Frag16 { v16h v; unsigned u[8]; };

// ------------------------------- GEMM ---------------------------------------
// C[M,N] = A[M,K] (f16, row-major, lda) * W[K,N] (f16, row-major, ldb) + bias
// optional relu; writes f32 and/or f16 outputs (row stride ldo).
// Block = 128 threads (4 waves). Block tile 64x64, wave tile 16x64, K step 32.
// Requires: lda, ldb multiples of 8 f16 whenever the fast path can trigger
// (true for every launch in this model except the N=3 bind head, which always
// takes the guarded path on the B side).
__global__ void k_gemm(const _Float16* __restrict__ A, int lda,
                       const _Float16* __restrict__ Bw, int ldb,
                       const float* __restrict__ bias,
                       float* outF, _Float16* outH,
                       int M, int N, int K, int ldo, int relu)
{
    __shared__ _Float16 As[64 * 32];       // A tile, row-major [row][k]
    __shared__ _Float16 Bs[64 * 32];       // B tile transposed: [n][k]
    const int tid  = threadIdx.x;
    const int wave = tid >> 5;
    const int lane = tid & 31;
    const int m0 = blockIdx.y * 64;
    const int n0 = blockIdx.x * 64;

    v8f c[4];
    #pragma unroll
    for (int t = 0; t < 4; ++t) c[t] = (v8f){0.f,0.f,0.f,0.f,0.f,0.f,0.f,0.f};

    for (int k0 = 0; k0 < K; k0 += 32) {
        // ---- prefetch next K tile (global_prefetch_b8), uniform guards ----
        if (m0 + 64 <= M && k0 + 64 <= K)
            __builtin_prefetch(&A[(size_t)(m0 + (tid >> 1)) * lda + k0 + 32 + ((tid & 1) << 4)], 0, 1);
        if (n0 + 64 <= N && k0 + 64 <= K && tid < 32)
            __builtin_prefetch(&Bw[(size_t)(k0 + 32 + tid) * ldb + n0], 0, 1);

        // ---- stage A tile (64 x 32 f16) ----
        if (m0 + 64 <= M && k0 + 32 <= K) {
            // fast path: 256 chunks of 8 f16; 2 per thread, b128 copies
            #pragma unroll
            for (int j = 0; j < 2; ++j) {
                int cch = tid + j * 128;
                int r = cch >> 2, ko = (cch & 3) << 3;
                int4 v = *(const int4*)&A[(size_t)(m0 + r) * lda + k0 + ko];
                *(int4*)&As[r * 32 + ko] = v;
            }
        } else {
            for (int i = tid; i < 64 * 16; i += 128) {
                int r = i >> 4, kk = (i & 15) << 1;
                int gm = m0 + r, gk = k0 + kk;
                unsigned v = 0u;
                if (gm < M) {
                    unsigned lo = (gk     < K) ? *(const unsigned short*)&A[(size_t)gm * lda + gk    ] : 0u;
                    unsigned hi = (gk + 1 < K) ? *(const unsigned short*)&A[(size_t)gm * lda + gk + 1] : 0u;
                    v = lo | (hi << 16);
                }
                ((unsigned*)As)[i] = v;
            }
        }

        // ---- stage B tile (32 x 64 f16) transposed into Bs[n][k] ----
        if (k0 + 32 <= K && n0 + 64 <= N) {
            // fast path: thread owns one (k-pair, n-chunk): 2x b128 loads,
            // pack (k,k+1) pairs, 8x ds_store_b32 into transposed layout.
            int p  = tid >> 3;              // k pair index 0..15 -> k = 2p
            int nc = (tid & 7) << 3;        // n chunk base 0,8,..,56
            int gk = k0 + (p << 1);
            int4 r0 = *(const int4*)&Bw[(size_t)gk * ldb + n0 + nc];
            int4 r1 = *(const int4*)&Bw[(size_t)(gk + 1) * ldb + n0 + nc];
            const unsigned short* a0 = (const unsigned short*)&r0;
            const unsigned short* a1 = (const unsigned short*)&r1;
            #pragma unroll
            for (int j = 0; j < 8; ++j) {
                unsigned pair = (unsigned)a0[j] | ((unsigned)a1[j] << 16);
                *(unsigned*)&Bs[(nc + j) * 32 + (p << 1)] = pair;
            }
        } else {
            for (int i = tid; i < 64 * 32; i += 128) {
                int n = i >> 5, kk = i & 31;
                int gk = k0 + kk, gn = n0 + n;
                _Float16 v = (_Float16)0.f;
                if (gk < K && gn < N) v = Bw[(size_t)gk * ldb + gn];
                Bs[n * 32 + kk] = v;
            }
        }
        __syncthreads();

        // ---- A fragment (16x32 f16, ISA 7.12.2 layout) ----
        const int arow  = (lane & 15) + wave * 16;
        const int khalf = (lane >> 4) << 3;       // 0 or 8
        Frag16 af;
        #pragma unroll
        for (int j = 0; j < 8; ++j) {
            int kk = ((j >> 2) << 4) + khalf + ((j & 3) << 1);
            af.u[j] = *(const unsigned*)&As[arow * 32 + kk];
        }
        // ---- 4 WMMAs across the 64-wide N strip ----
        const int kb = (lane >> 4) << 4;          // 0 or 16
        #pragma unroll
        for (int t = 0; t < 4; ++t) {
            Frag16 bf;
            const int col = (lane & 15) + t * 16;
            #pragma unroll
            for (int j = 0; j < 8; ++j)
                bf.u[j] = *(const unsigned*)&Bs[col * 32 + ((j >> 2) << 4) + khalf + ((j & 3) << 1)];
            c[t] = __builtin_amdgcn_wmma_f32_16x16x32_f16(
                       false, af.v, false, bf.v, (short)0, c[t], false, false);
        }
        __syncthreads();
    }

    // ---- epilogue: C layout VGPR r -> M=r (lanes 0-15) / M=r+8 (lanes 16-31)
    const int mbase = m0 + wave * 16 + ((lane >> 4) << 3);
    #pragma unroll
    for (int t = 0; t < 4; ++t) {
        const int gn = n0 + t * 16 + (lane & 15);
        const float bv = (bias && gn < N) ? bias[gn] : 0.f;
        #pragma unroll
        for (int r = 0; r < 8; ++r) {
            const int gm = mbase + r;
            if (gm < M && gn < N) {
                float v = c[t][r] + bv;
                if (relu) v = fmaxf(v, 0.f);
                if (outF) outF[(size_t)gm * ldo + gn] = v;
                if (outH) outH[(size_t)gm * ldo + gn] = (_Float16)v;
            }
        }
    }
}

// ----------------------------- helpers --------------------------------------
__global__ void k_f2h(const float* __restrict__ in, _Float16* __restrict__ out, size_t n)
{
    size_t i = (size_t)blockIdx.x * blockDim.x + threadIdx.x;
    if (i < n) out[i] = (_Float16)in[i];
}

__global__ void k_zero_f(float* p, size_t n)
{
    size_t i = (size_t)blockIdx.x * blockDim.x + threadIdx.x;
    if (i < n) p[i] = 0.f;
}

// Mcat[e] = [ h[dst[e]] , h[src[e]] , edge_attr[e] ]   (E x 544, f16)
__global__ void k_gather_msg(const _Float16* __restrict__ hH,
                             const float* __restrict__ eattr,
                             const int* __restrict__ ei,
                             _Float16* __restrict__ Mcat)
{
    size_t i = (size_t)blockIdx.x * blockDim.x + threadIdx.x;
    if (i >= (size_t)N_EDGES * 544) return;
    int e = (int)(i / 544), c = (int)(i % 544);
    _Float16 v;
    if (c < EMB)            { int d = ei[N_EDGES + e]; v = hH[(size_t)d * EMB + c]; }
    else if (c < 2 * EMB)   { int s = ei[e];           v = hH[(size_t)s * EMB + (c - EMB)]; }
    else                    { v = (_Float16)eattr[(size_t)e * DE + (c - 2 * EMB)]; }
    Mcat[i] = v;
}

// column stats for BN: stats[0..255]=sum, stats[256..511]=sumsq (pre-zeroed)
__global__ void k_bn_stats(const float* __restrict__ y, float* __restrict__ stats, int R)
{
    const int col = threadIdx.x;                 // blockDim.x == 256
    const int r0  = blockIdx.x * 256;
    float s = 0.f, sq = 0.f;
    for (int r = r0; r < r0 + 256; ++r) {
        if (r < R) { float v = y[(size_t)r * EMB + col]; s += v; sq += v * v; }
    }
    atomicAdd(&stats[col], s);
    atomicAdd(&stats[EMB + col], sq);
}

// out = relu( (y-mean)*rsqrt(var+eps)*g + b ), biased batch var
__global__ void k_bn_apply(const float* __restrict__ y, const float* __restrict__ stats,
                           const float* __restrict__ gamma, const float* __restrict__ beta,
                           _Float16* outH, float* outF, int R)
{
    size_t i = (size_t)blockIdx.x * blockDim.x + threadIdx.x;
    if (i >= (size_t)R * EMB) return;
    int col = (int)(i & (EMB - 1));
    float invR = 1.f / (float)R;
    float m  = stats[col] * invR;
    float va = stats[EMB + col] * invR - m * m;
    float v  = (y[i] - m) * rsqrtf(va + 1e-5f) * gamma[col] + beta[col];
    v = fmaxf(v, 0.f);
    if (outH) outH[i] = (_Float16)v;
    if (outF) outF[i] = v;
}

// aggr[dst[e]] += m[e]   (fp32 atomics)
__global__ void k_scatter_add(const _Float16* __restrict__ mH, const int* __restrict__ ei,
                              float* __restrict__ aggr)
{
    size_t i = (size_t)blockIdx.x * blockDim.x + threadIdx.x;
    if (i >= (size_t)N_EDGES * EMB) return;
    int e = (int)(i >> 8), c = (int)(i & (EMB - 1));
    int d = ei[N_EDGES + e];
    atomicAdd(&aggr[(size_t)d * EMB + c], (float)mH[i]);
}

// Ucat[n] = [ h[n] , aggr[n] ]   (N x 512, f16)
__global__ void k_concat_upd(const _Float16* __restrict__ hH, const float* __restrict__ aggr,
                             _Float16* __restrict__ Ucat)
{
    size_t i = (size_t)blockIdx.x * blockDim.x + threadIdx.x;
    if (i >= (size_t)N_NODES * 2 * EMB) return;
    int n = (int)(i >> 9), c = (int)(i & (2 * EMB - 1));
    Ucat[i] = (c < EMB) ? hH[(size_t)n * EMB + c]
                        : (_Float16)aggr[(size_t)n * EMB + (c - EMB)];
}

// h += u (residual), kept in f16 master
__global__ void k_residual(_Float16* __restrict__ hH, const float* __restrict__ u)
{
    size_t i = (size_t)blockIdx.x * blockDim.x + threadIdx.x;
    if (i >= (size_t)N_NODES * EMB) return;
    hH[i] = (_Float16)((float)hH[i] + u[i]);
}

__global__ void k_pool_sum(const _Float16* __restrict__ hH, const int* __restrict__ batch,
                           float* __restrict__ sums)
{
    size_t i = (size_t)blockIdx.x * blockDim.x + threadIdx.x;
    if (i >= (size_t)N_NODES * EMB) return;
    int n = (int)(i >> 8), c = (int)(i & (EMB - 1));
    atomicAdd(&sums[(size_t)batch[n] * EMB + c], (float)hH[i]);
}

__global__ void k_pool_cnt(const int* __restrict__ batch, float* __restrict__ cnt)
{
    int i = blockIdx.x * blockDim.x + threadIdx.x;
    if (i < N_NODES) atomicAdd(&cnt[batch[i]], 1.f);
}

__global__ void k_pool_div(const float* __restrict__ sums, const float* __restrict__ cnt,
                           _Float16* __restrict__ g0H)
{
    size_t i = (size_t)blockIdx.x * blockDim.x + threadIdx.x;
    if (i >= (size_t)NB * EMB) return;
    g0H[i] = (_Float16)(sums[i] / fmaxf(cnt[i >> 8], 1.f));
}

// per-sample outer-product attention:
//   energy[d,e] = Q[d]*K[e]/8 ; softmax over e ; o[d] = sum_e a * V[e]
// output flattened torch-style: col = d*NH + head
__global__ void k_attn(const float* __restrict__ Q, const float* __restrict__ K,
                       const float* __restrict__ V, _Float16* __restrict__ oH)
{
    int idx = blockIdx.x * blockDim.x + threadIdx.x;     // B*HID threads
    if (idx >= NB * HIDD) return;
    int b = idx / HIDD, r = idx % HIDD;
    int head = r / HD, d = r % HD;
    const float qd = Q[(size_t)b * HIDD + head * HD + d];
    const float* kk = K + (size_t)b * HIDD + head * HD;
    const float* vv = V + (size_t)b * HIDD + head * HD;
    const float scale = 0.125f;                          // 1/sqrt(64)
    float mx = -1e30f;
    for (int e = 0; e < HD; ++e) mx = fmaxf(mx, qd * kk[e] * scale);
    float s = 0.f, acc = 0.f;
    for (int e = 0; e < HD; ++e) {
        float w = __expf(qd * kk[e] * scale - mx);
        s += w; acc += w * vv[e];
    }
    oH[(size_t)b * HIDD + d * NH + head] = (_Float16)(acc / s);
}

// out = a + b, emit f16
__global__ void k_add_h(const float* __restrict__ a, const float* __restrict__ b,
                        _Float16* __restrict__ outH, size_t n)
{
    size_t i = (size_t)blockIdx.x * blockDim.x + threadIdx.x;
    if (i < n) outH[i] = (_Float16)(a[i] + b[i]);
}

// BCE-with-logits mean + sigmoid; out[0]=loss, out[1..B*3]=sigmoid
__global__ void k_loss(const float* __restrict__ bind, const float* __restrict__ target,
                       float* __restrict__ out)
{
    __shared__ float red[1024];
    int t = threadIdx.x;
    float acc = 0.f;
    for (int i = t; i < NB * 3; i += 1024) {
        float z = bind[i], tg = target[i];
        acc += fmaxf(z, 0.f) - z * tg + log1pf(__expf(-fabsf(z)));
        out[1 + i] = 1.f / (1.f + __expf(-z));
    }
    red[t] = acc; __syncthreads();
    for (int s = 512; s > 0; s >>= 1) { if (t < s) red[t] += red[t + s]; __syncthreads(); }
    if (t == 0) out[0] = red[0] / (float)(NB * 3);
}

// ----------------------------- host side ------------------------------------
static inline void* wsa(char* base, size_t& off, size_t bytes)
{
    off = (off + 255) & ~(size_t)255;
    void* p = base + off;
    off += bytes;
    return p;
}

static void gemm(hipStream_t s, const _Float16* A, int lda, const _Float16* W, int ldb,
                 const float* bias, float* outF, _Float16* outH,
                 int M, int N, int K, int ldo, int relu)
{
    dim3 g((N + 63) / 64, (M + 63) / 64), b(128);
    k_gemm<<<g, b, 0, s>>>(A, lda, W, ldb, bias, outF, outH, M, N, K, ldo, relu);
}

extern "C" void kernel_launch(void* const* d_in, const int* in_sizes, int n_in,
                              void* d_out, int out_size, void* d_ws, size_t ws_size,
                              hipStream_t stream)
{
    (void)in_sizes; (void)n_in; (void)out_size; (void)ws_size;
    const float* x       = (const float*)d_in[0];
    const int*   ei      = (const int*)  d_in[1];
    const float* eattr   = (const float*)d_in[2];
    const int*   batch   = (const int*)  d_in[3];
    const float* fp      = (const float*)d_in[4];
    const float* target  = (const float*)d_in[5];
    const float* lin_w   = (const float*)d_in[6];
    const float* lin_b   = (const float*)d_in[7];
    const float* msg_w1  = (const float*)d_in[8];
    const float* msg_b1  = (const float*)d_in[9];
    const float* msg_g1  = (const float*)d_in[10];
    const float* msg_be1 = (const float*)d_in[11];
    const float* msg_w2  = (const float*)d_in[12];
    const float* msg_b2  = (const float*)d_in[13];
    const float* msg_g2  = (const float*)d_in[14];
    const float* msg_be2 = (const float*)d_in[15];
    const float* upd_w1  = (const float*)d_in[16];
    const float* upd_b1  = (const float*)d_in[17];
    const float* upd_g1  = (const float*)d_in[18];
    const float* upd_be1 = (const float*)d_in[19];
    const float* upd_w2  = (const float*)d_in[20];
    const float* upd_b2  = (const float*)d_in[21];
    const float* upd_g2  = (const float*)d_in[22];
    const float* upd_be2 = (const float*)d_in[23];
    const float* gl_w1   = (const float*)d_in[24];
    const float* gl_b1   = (const float*)d_in[25];
    const float* gl_w2   = (const float*)d_in[26];
    const float* gl_b2   = (const float*)d_in[27];
    const float* gl_w3   = (const float*)d_in[28];
    const float* gl_b3   = (const float*)d_in[29];
    const float* fp_w    = (const float*)d_in[30];
    const float* fp_b    = (const float*)d_in[31];
    const float* att_w   = (const float*)d_in[32];
    const float* att_b   = (const float*)d_in[33];
    const float* bind_w  = (const float*)d_in[34];
    const float* bind_b  = (const float*)d_in[35];

    char* base = (char*)d_ws;
    size_t off = 0;
    auto AH = [&](size_t elems) { return (_Float16*)wsa(base, off, elems * sizeof(_Float16)); };
    auto AF = [&](size_t elems) { return (float*)   wsa(base, off, elems * sizeof(float)); };

    // ---- f16 weight mirrors ----
    _Float16* lin_wH  = AH((size_t)DIN * EMB);
    _Float16* msgw1H  = AH((size_t)NL * (2 * EMB + DE) * EMB);
    _Float16* msgw2H  = AH((size_t)NL * EMB * EMB);
    _Float16* updw1H  = AH((size_t)NL * 2 * EMB * EMB);
    _Float16* updw2H  = AH((size_t)NL * EMB * EMB);
    _Float16* glw1H   = AH((size_t)EMB * 1024);
    _Float16* glw2H   = AH((size_t)1024 * 1024);
    _Float16* glw3H   = AH((size_t)1024 * HIDD);
    _Float16* fpwH    = AH((size_t)2048 * HIDD);
    _Float16* attwH   = AH((size_t)2 * 4 * HIDD * HIDD);
    _Float16* bindwH  = AH((size_t)HIDD * 3);
    // ---- activations ----
    _Float16* xH    = AH((size_t)N_NODES * DIN);
    _Float16* hH    = AH((size_t)N_NODES * EMB);
    _Float16* bigA  = AH((size_t)N_EDGES * (2 * EMB + DE));   // Mcat / Ucat
    float*    yF    = AF((size_t)N_EDGES * EMB);              // GEMM pre-activations
    _Float16* actH  = AH((size_t)N_EDGES * EMB);              // BN/relu outputs (f16)
    float*    aggr  = AF((size_t)N_NODES * EMB);
    float*    stats = AF(2 * EMB);
    float*    sums  = AF((size_t)NB * EMB);
    float*    cnt   = AF(NB);
    _Float16* g0H   = AH((size_t)NB * EMB);
    _Float16* t1H   = AH((size_t)NB * 1024);
    _Float16* t2H   = AH((size_t)NB * 1024);
    float*    gF    = AF((size_t)NB * HIDD);
    _Float16* gH    = AH((size_t)NB * HIDD);
    _Float16* fpH   = AH((size_t)NB * 2048);
    _Float16* fpfH  = AH((size_t)NB * HIDD);
    float*    Qf    = AF((size_t)NB * HIDD);
    float*    Kf    = AF((size_t)NB * HIDD);
    float*    Vf    = AF((size_t)NB * HIDD);
    _Float16* oH    = AH((size_t)NB * HIDD);
    float*    attF  = AF((size_t)NB * HIDD);
    _Float16* g2H   = AH((size_t)NB * HIDD);
    _Float16* g3H   = AH((size_t)NB * HIDD);
    float*    bindF = AF((size_t)NB * 3);

    auto cvt = [&](const float* src, _Float16* dst, size_t n) {
        k_f2h<<<(unsigned)((n + 255) / 256), 256, 0, stream>>>(src, dst, n);
    };
    auto zero = [&](float* p, size_t n) {
        k_zero_f<<<(unsigned)((n + 255) / 256), 256, 0, stream>>>(p, n);
    };
    auto bn = [&](const float* y, int R, const float* g, const float* be,
                  _Float16* outH, float* outF) {
        zero(stats, 2 * EMB);
        k_bn_stats<<<(R + 255) / 256, 256, 0, stream>>>(y, stats, R);
        size_t n = (size_t)R * EMB;
        k_bn_apply<<<(unsigned)((n + 255) / 256), 256, 0, stream>>>(y, stats, g, be, outH, outF, R);
    };

    // ---- convert weights + inputs to f16 ----
    cvt(lin_w,  lin_wH, (size_t)DIN * EMB);
    cvt(msg_w1, msgw1H, (size_t)NL * (2 * EMB + DE) * EMB);
    cvt(msg_w2, msgw2H, (size_t)NL * EMB * EMB);
    cvt(upd_w1, updw1H, (size_t)NL * 2 * EMB * EMB);
    cvt(upd_w2, updw2H, (size_t)NL * EMB * EMB);
    cvt(gl_w1,  glw1H,  (size_t)EMB * 1024);
    cvt(gl_w2,  glw2H,  (size_t)1024 * 1024);
    cvt(gl_w3,  glw3H,  (size_t)1024 * HIDD);
    cvt(fp_w,   fpwH,   (size_t)2048 * HIDD);
    cvt(att_w,  attwH,  (size_t)2 * 4 * HIDD * HIDD);
    cvt(bind_w, bindwH, (size_t)HIDD * 3);
    cvt(x,      xH,     (size_t)N_NODES * DIN);
    cvt(fp,     fpH,    (size_t)NB * 2048);

    // ---- h = x @ lin_w + lin_b ----
    gemm(stream, xH, DIN, lin_wH, EMB, lin_b, nullptr, hH, N_NODES, EMB, DIN, EMB, 0);

    // ---- MPNN layers ----
    for (int l = 0; l < NL; ++l) {
        const int KM = 2 * EMB + DE;  // 544
        size_t nM = (size_t)N_EDGES * KM;
        k_gather_msg<<<(unsigned)((nM + 255) / 256), 256, 0, stream>>>(hH, eattr, ei, bigA);
        gemm(stream, bigA, KM, msgw1H + (size_t)l * KM * EMB, EMB,
             msg_b1 + l * EMB, yF, nullptr, N_EDGES, EMB, KM, EMB, 0);
        bn(yF, N_EDGES, msg_g1 + l * EMB, msg_be1 + l * EMB, actH, nullptr);
        gemm(stream, actH, EMB, msgw2H + (size_t)l * EMB * EMB, EMB,
             msg_b2 + l * EMB, yF, nullptr, N_EDGES, EMB, EMB, EMB, 0);
        bn(yF, N_EDGES, msg_g2 + l * EMB, msg_be2 + l * EMB, actH, nullptr);
        // aggregate
        zero(aggr, (size_t)N_NODES * EMB);
        size_t nS = (size_t)N_EDGES * EMB;
        k_scatter_add<<<(unsigned)((nS + 255) / 256), 256, 0, stream>>>(actH, ei, aggr);
        // update
        size_t nU = (size_t)N_NODES * 2 * EMB;
        k_concat_upd<<<(unsigned)((nU + 255) / 256), 256, 0, stream>>>(hH, aggr, bigA);
        gemm(stream, bigA, 2 * EMB, updw1H + (size_t)l * 2 * EMB * EMB, EMB,
             upd_b1 + l * EMB, yF, nullptr, N_NODES, EMB, 2 * EMB, EMB, 0);
        bn(yF, N_NODES, upd_g1 + l * EMB, upd_be1 + l * EMB, actH, nullptr);
        gemm(stream, actH, EMB, updw2H + (size_t)l * EMB * EMB, EMB,
             upd_b2 + l * EMB, yF, nullptr, N_NODES, EMB, EMB, EMB, 0);
        bn(yF, N_NODES, upd_g2 + l * EMB, upd_be2 + l * EMB, nullptr, yF);  // in-place f32
        size_t nR = (size_t)N_NODES * EMB;
        k_residual<<<(unsigned)((nR + 255) / 256), 256, 0, stream>>>(hH, yF);
    }

    // ---- global mean pool ----
    zero(sums, (size_t)NB * EMB);
    zero(cnt, NB);
    size_t nP = (size_t)N_NODES * EMB;
    k_pool_sum<<<(unsigned)((nP + 255) / 256), 256, 0, stream>>>(hH, batch, sums);
    k_pool_cnt<<<(N_NODES + 255) / 256, 256, 0, stream>>>(batch, cnt);
    k_pool_div<<<(NB * EMB + 255) / 256, 256, 0, stream>>>(sums, cnt, g0H);

    // ---- graph MLP + fp projection ----
    gemm(stream, g0H, EMB, glw1H, 1024, gl_b1, nullptr, t1H, NB, 1024, EMB, 1024, 1);
    gemm(stream, t1H, 1024, glw2H, 1024, gl_b2, nullptr, t2H, NB, 1024, 1024, 1024, 1);
    gemm(stream, t2H, 1024, glw3H, HIDD, gl_b3, gF, gH, NB, HIDD, 1024, HIDD, 0);
    gemm(stream, fpH, 2048, fpwH, HIDD, fp_b, nullptr, fpfH, NB, HIDD, 2048, HIDD, 1);

    // ---- cross attention: fp -> graph, with residual ----
    const size_t AW = (size_t)HIDD * HIDD;
    gemm(stream, fpfH, HIDD, attwH + 0 * AW, HIDD, att_b + 0 * HIDD, Qf, nullptr, NB, HIDD, HIDD, HIDD, 0);
    gemm(stream, gH,   HIDD, attwH + 1 * AW, HIDD, att_b + 1 * HIDD, Kf, nullptr, NB, HIDD, HIDD, HIDD, 0);
    gemm(stream, gH,   HIDD, attwH + 2 * AW, HIDD, att_b + 2 * HIDD, Vf, nullptr, NB, HIDD, HIDD, HIDD, 0);
    k_attn<<<(NB * HIDD + 255) / 256, 256, 0, stream>>>(Qf, Kf, Vf, oH);
    gemm(stream, oH, HIDD, attwH + 3 * AW, HIDD, att_b + 3 * HIDD, attF, nullptr, NB, HIDD, HIDD, HIDD, 0);
    k_add_h<<<(NB * HIDD + 255) / 256, 256, 0, stream>>>(gF, attF, g2H, (size_t)NB * HIDD);

    // ---- self attention (no residual) ----
    gemm(stream, g2H, HIDD, attwH + 4 * AW, HIDD, att_b + 4 * HIDD, Qf, nullptr, NB, HIDD, HIDD, HIDD, 0);
    gemm(stream, g2H, HIDD, attwH + 5 * AW, HIDD, att_b + 5 * HIDD, Kf, nullptr, NB, HIDD, HIDD, HIDD, 0);
    gemm(stream, g2H, HIDD, attwH + 6 * AW, HIDD, att_b + 6 * HIDD, Vf, nullptr, NB, HIDD, HIDD, HIDD, 0);
    k_attn<<<(NB * HIDD + 255) / 256, 256, 0, stream>>>(Qf, Kf, Vf, oH);
    gemm(stream, oH, HIDD, attwH + 7 * AW, HIDD, att_b + 7 * HIDD, attF, nullptr, NB, HIDD, HIDD, HIDD, 0);
    cvt(attF, g3H, (size_t)NB * HIDD);

    // ---- bind + loss ----
    gemm(stream, g3H, HIDD, bindwH, 3, bind_b, bindF, nullptr, NB, 3, HIDD, 3, 0);
    k_loss<<<1, 1024, 0, stream>>>(bindF, target, (float*)d_out);
}
